// SkeletonEncoder_80315888435268
// MI455X (gfx1250) — matrix operand order repaired and checked by reading, not compiled, not run on hardware
//
#include <hip/hip_runtime.h>

typedef __attribute__((ext_vector_type(16))) _Float16 v16h;
typedef __attribute__((ext_vector_type(8)))  _Float16 v8h;
typedef __attribute__((ext_vector_type(4)))  _Float16 v4h;
typedef __attribute__((ext_vector_type(8)))  float    v8f;
typedef __attribute__((ext_vector_type(4)))  float    v4f;

#define D_MODEL  128
#define D_INNER  256
#define D_STATE  32
#define DT_RANK  8
#define SEQ_T    4096
#define NBATCH   16
#define MROWS    (NBATCH * SEQ_T)   // 65536 token rows

// --------------------------------------------------------------------------
// Weight convert f32 -> f16, zero-padded AND TRANSPOSED: dstT[NP][KP]
// so that B fragments are contiguous along K for the WMMA lane layout.
// --------------------------------------------------------------------------
__global__ void k_convert_pad_T(const float* __restrict__ src, _Float16* __restrict__ dstT,
                                int K, int N, int KP, int NP) {
  int i = blockIdx.x * blockDim.x + threadIdx.x;
  if (i >= KP * NP) return;
  int n = i / KP, k = i - n * KP;
  float v = (k < K && n < N) ? src[k * N + n] : 0.f;
  dstT[i] = (_Float16)v;
}

// --------------------------------------------------------------------------
// WMMA GEMM: out[M][NPAD] = act[M][KREAL](f32) x w (as wT[NPAD][KPAD] f16)
// Block = 128 threads (4 waves). Block tile 32 rows x 64 cols.
// Each wave owns two 16x16 M-subtiles sharing one B fragment per k-step
// (2 v_wmma per B load). A strip in LDS (f16, vectorized fill).
// Epilogue: LDS-staged accumulators -> vectorized coalesced stores; the
// partial-N guard is compile-time dead when NPAD % 64 == 0.
// --------------------------------------------------------------------------
template<int KREAL, int KPAD, int NPAD>
__global__ void k_gemm(const float* __restrict__ act,
                       const _Float16* __restrict__ wT,
                       float* out, const float* residual) {
  __shared__ _Float16 aT[32 * KPAD];
  __shared__ float    cT[32 * 64];
  const int tid  = threadIdx.x;
  const int wave = tid >> 5;
  const int lane = tid & 31;
  const size_t row0 = (size_t)blockIdx.x * 32;
  const int colBase = blockIdx.y * 64;
  const int col0 = colBase + wave * 16;

  // cooperative A-strip load (4 f32 -> 4 f16 per step; KREAL % 4 == 0)
  for (int i = tid * 4; i < 32 * KPAD; i += 128 * 4) {
    int rr = i / KPAD, kk = i - rr * KPAD;
    v4h ah;
    if (KREAL == KPAD || kk < KREAL) {
      v4f av = *(const v4f*)(act + (row0 + rr) * (size_t)KREAL + kk);
#pragma unroll
      for (int e = 0; e < 4; ++e) ah[e] = (_Float16)av[e];
    } else {
      ah = (v4h)(_Float16)0.f;
    }
    *(v4h*)(aT + rr * KPAD + kk) = ah;
  }
  __syncthreads();

  const bool active = (col0 + 16) <= NPAD;   // wave-uniform partial-N guard

  if (active) {
    // fragment index maps (wave32): contiguous K order k = 16*(lane/16) + s
    const int mA  = lane & 15;
    const int kA0 = (lane < 16) ? 0 : 16;    // elements 0..7
    const int kA1 = (lane < 16) ? 8 : 24;    // elements 8..15
    const int nB  = col0 + (lane & 15);
    const int kB  = (lane < 16) ? 0 : 16;
    const _Float16* bBase = wT + (size_t)nB * KPAD + kB;

    v8f acc0 = {}, acc1 = {};
#pragma unroll
    for (int k0 = 0; k0 < KPAD; k0 += 32) {
      const _Float16* ap0 = aT + mA * KPAD + k0;          // rows 0..15
      const _Float16* ap1 = aT + (16 + mA) * KPAD + k0;   // rows 16..31
      v16h a0 = __builtin_shufflevector(*(const v8h*)(ap0 + kA0),
                                        *(const v8h*)(ap0 + kA1),
                 0, 1, 2, 3, 4, 5, 6, 7, 8, 9, 10, 11, 12, 13, 14, 15);
      v16h a1 = __builtin_shufflevector(*(const v8h*)(ap1 + kA0),
                                        *(const v8h*)(ap1 + kA1),
                 0, 1, 2, 3, 4, 5, 6, 7, 8, 9, 10, 11, 12, 13, 14, 15);
      v16h b = *(const v16h*)(bBase + k0);   // contiguous K (transposed weights)
      acc0 = __builtin_amdgcn_wmma_f32_16x16x32_f16(false, a0, false, b,
                                                    (short)0, acc0, false, false);
      acc1 = __builtin_amdgcn_wmma_f32_16x16x32_f16(false, a1, false, b,
                                                    (short)0, acc1, false, false);
    }

    // park wave tiles in LDS for coalesced epilogue
    const int mBase = (lane < 16) ? 0 : 8;
    const int nLoc  = wave * 16 + (lane & 15);
#pragma unroll
    for (int j = 0; j < 8; ++j) {
      cT[(mBase + j) * 64 + nLoc]        = acc0[j];
      cT[(16 + mBase + j) * 64 + nLoc]   = acc1[j];
    }
  }
  __syncthreads();

  // vectorized coalesced store (+ optional residual, uniform branches)
  constexpr bool fullN = (NPAD % 64 == 0);
  const int cols = fullN ? 64 : ((NPAD - colBase < 64) ? NPAD - colBase : 64);
  if (residual) {
#pragma unroll
    for (int i = tid * 4; i < 32 * 64; i += 128 * 4) {
      int r = i >> 6, cc = i & 63;
      if (fullN || cc < cols) {
        size_t idx = (row0 + r) * (size_t)NPAD + colBase + cc;
        v4f v = *(const v4f*)(cT + i);
        v4f rv = *(const v4f*)(residual + idx);
        *(v4f*)(out + idx) = v + rv;
      }
    }
  } else {
#pragma unroll
    for (int i = tid * 4; i < 32 * 64; i += 128 * 4) {
      int r = i >> 6, cc = i & 63;
      if (fullN || cc < cols) {
        size_t idx = (row0 + r) * (size_t)NPAD + colBase + cc;
        *(v4f*)(out + idx) = *(const v4f*)(cT + i);
      }
    }
  }
}

// --------------------------------------------------------------------------
// LayerNorm (+ fe bias) + ReLU over rows of 128. One block (128 thr) per row.
// --------------------------------------------------------------------------
__global__ void k_ln_relu(const float* __restrict__ hraw, const float* __restrict__ fe_b,
                          const float* __restrict__ g, const float* __restrict__ bb,
                          float* __restrict__ out) {
  const size_t m = blockIdx.x;
  const int c = threadIdx.x;
  float v = hraw[m * 128 + c] + fe_b[c];
  float s = v, q = v * v;
#pragma unroll
  for (int off = 16; off > 0; off >>= 1) {
    s += __shfl_down(s, off, 32);
    q += __shfl_down(q, off, 32);
  }
  __shared__ float rs[4], rq[4];
  const int w = c >> 5, lane = c & 31;
  if (lane == 0) { rs[w] = s; rq[w] = q; }
  __syncthreads();
  float S = rs[0] + rs[1] + rs[2] + rs[3];
  float Q = rq[0] + rq[1] + rq[2] + rq[3];
  float mu  = S * (1.f / 128.f);
  float var = Q * (1.f / 128.f) - mu * mu;
  float y = (v - mu) * rsqrtf(var + 1e-5f) * g[c] + bb[c];
  out[m * 128 + c] = y > 0.f ? y : 0.f;
}

// --------------------------------------------------------------------------
// RMSNorm over rows of 128 (in-place safe). One block (128 thr) per row.
// --------------------------------------------------------------------------
__global__ void k_rmsnorm(const float* __restrict__ in, const float* __restrict__ w,
                          float* out) {
  const size_t m = blockIdx.x;
  const int c = threadIdx.x;
  float v = in[m * 128 + c];
  float q = v * v;
#pragma unroll
  for (int off = 16; off > 0; off >>= 1) q += __shfl_down(q, off, 32);
  __shared__ float rq[4];
  const int wv = c >> 5, lane = c & 31;
  if (lane == 0) rq[wv] = q;
  __syncthreads();
  float Q = rq[0] + rq[1] + rq[2] + rq[3];
  out[m * 128 + c] = v * rsqrtf(Q * (1.f / 128.f) + 1e-6f) * w[c];
}

// --------------------------------------------------------------------------
// Causal depthwise conv (K=4) over xc = xz[:, :256], + SiLU
// --------------------------------------------------------------------------
__global__ void k_conv_silu(const float* __restrict__ xz, const float* __restrict__ cw,
                            const float* __restrict__ cb, float* __restrict__ xc) {
  size_t idx = (size_t)blockIdx.x * blockDim.x + threadIdx.x;
  if (idx >= (size_t)MROWS * D_INNER) return;
  int d = (int)(idx & (D_INNER - 1));
  size_t m = idx >> 8;
  int t = (int)(m & (SEQ_T - 1));
  float acc = cb[d];
#pragma unroll
  for (int k = 0; k < 4; ++k) {
    int tt = t - 3 + k;
    if (tt >= 0) acc += xz[(m - 3 + k) * 512 + d] * cw[d * 4 + k];
  }
  xc[idx] = acc / (1.f + expf(-acc));  // silu
}

// --------------------------------------------------------------------------
// dt = softplus(dbl[:, :8] @ dt_proj_w + dt_proj_b)   (K=8: plain VALU)
// --------------------------------------------------------------------------
__global__ void k_dt(const float* __restrict__ dbl, const float* __restrict__ dw,
                     const float* __restrict__ db, float* __restrict__ dt) {
  size_t idx = (size_t)blockIdx.x * blockDim.x + threadIdx.x;
  if (idx >= (size_t)MROWS * D_INNER) return;
  int d = (int)(idx & (D_INNER - 1));
  size_t m = idx >> 8;
  float acc = db[d];
#pragma unroll
  for (int r = 0; r < DT_RANK; ++r) acc += dbl[m * 80 + r] * dw[r * D_INNER + d];
  dt[idx] = (acc > 20.f) ? acc : log1pf(expf(acc));
}

// --------------------------------------------------------------------------
// Sequential selective scan. One block per batch, one lane per channel d.
// 32 hst + 32 pst + 32 A kept in VGPRs; B/C staged via LDS each step.
// y written in-place over the dt buffer; includes +u*D and SiLU(z) gating.
// --------------------------------------------------------------------------
__global__ void k_scan(float* __restrict__ dtY, const float* __restrict__ xc,
                       const float* __restrict__ dbl, const float* __restrict__ xz,
                       const float* __restrict__ A_log, const float* __restrict__ Dp,
                       const float* __restrict__ alpha_p, const float* __restrict__ beta_p) {
  const int b = blockIdx.x;
  const int d = threadIdx.x;
  float A_[D_STATE], hst[D_STATE], pst[D_STATE];
#pragma unroll
  for (int s = 0; s < D_STATE; ++s) {
    A_[s] = -expf(A_log[d * D_STATE + s]);
    hst[s] = 0.f;
    pst[s] = 0.f;
  }
  const float alpha = alpha_p[0], beta = beta_p[0], Dd = Dp[d];
  __shared__ float BC[2 * D_STATE];

  for (int t = 0; t < SEQ_T; ++t) {
    size_t m = (size_t)b * SEQ_T + t;
    __syncthreads();                         // protect previous step's BC reads
    if (d < 2 * D_STATE) BC[d] = dbl[m * 80 + DT_RANK + d];
    if (t + 1 < SEQ_T)
      __builtin_prefetch(&dbl[(m + 1) * 80 + DT_RANK], 0, 0);
    __syncthreads();

    float dt  = dtY[m * D_INNER + d];
    float u   = xc[m * D_INNER + d];
    float dtu = dt * u;
    float y = 0.f;
#pragma unroll
    for (int s = 0; s < D_STATE; ++s) {
      float dA = __expf(dt * A_[s]);
      pst[s] = beta * pst[s] + dtu * BC[s];
      hst[s] = dA * hst[s] + alpha * pst[s];
      y += hst[s] * BC[D_STATE + s];
    }
    float z = xz[m * 512 + D_INNER + d];
    y = y + u * Dd;
    y = y * (z / (1.f + expf(-z)));
    dtY[m * D_INNER + d] = y;
  }
}

// --------------------------------------------------------------------------
// Host side
// --------------------------------------------------------------------------
extern "C" void kernel_launch(void* const* d_in, const int* in_sizes, int n_in,
                              void* d_out, int out_size, void* d_ws, size_t ws_size,
                              hipStream_t stream) {
  (void)in_sizes; (void)n_in; (void)out_size; (void)ws_size;

  const float* x            = (const float*)d_in[0];
  const float* fe_w         = (const float*)d_in[1];
  const float* fe_b         = (const float*)d_in[2];
  const float* ln_g         = (const float*)d_in[3];
  const float* ln_b         = (const float*)d_in[4];
  const float* final_norm_w = (const float*)d_in[5];
  // per-layer params: norm_w, in_proj_w, conv_w, conv_b, x_proj_w,
  //                   dt_proj_w, dt_proj_b, A_log, D, out_proj_w, alpha, beta
  auto LP = [&](int l, int j) { return (const float*)d_in[6 + 12 * l + j]; };

  // workspace carve-up
  char* ws = (char*)d_ws;
  size_t off = 0;
  auto carve = [&](size_t bytes) -> void* {
    void* p = ws + off;
    off += (bytes + 255) & ~(size_t)255;
    return p;
  };
  float* xz  = (float*)carve((size_t)MROWS * 512 * sizeof(float)); // in_proj out (also fe scratch)
  float* xc  = (float*)carve((size_t)MROWS * 256 * sizeof(float)); // conv+silu out
  float* dtY = (float*)carve((size_t)MROWS * 256 * sizeof(float)); // rmsnorm r / dt / y (reused)
  float* dbl = (float*)carve((size_t)MROWS * 80  * sizeof(float)); // x_proj out (padded 72->80)
  _Float16* wfe = (_Float16*)carve(128 * 64 * sizeof(_Float16));   // transposed [N][K]
  _Float16* wip[2], *wxp[2], *wop[2];
  for (int l = 0; l < 2; ++l) {
    wip[l] = (_Float16*)carve(512 * 128 * sizeof(_Float16));
    wxp[l] = (_Float16*)carve(80  * 256 * sizeof(_Float16));
    wop[l] = (_Float16*)carve(128 * 256 * sizeof(_Float16));
  }

  float* h = (float*)d_out;   // h lives in d_out; final rmsnorm is in-place

  // ---- convert weights to transposed padded f16 ----
  auto cgrid = [](int n) { return dim3((n + 255) / 256); };
  k_convert_pad_T<<<cgrid(64 * 128), 256, 0, stream>>>(fe_w, wfe, 60, 128, 64, 128);
  for (int l = 0; l < 2; ++l) {
    k_convert_pad_T<<<cgrid(128 * 512), 256, 0, stream>>>(LP(l, 1), wip[l], 128, 512, 128, 512);
    k_convert_pad_T<<<cgrid(256 * 80),  256, 0, stream>>>(LP(l, 4), wxp[l], 256, 72, 256, 80);
    k_convert_pad_T<<<cgrid(256 * 128), 256, 0, stream>>>(LP(l, 9), wop[l], 256, 128, 256, 128);
  }

  const dim3 gB(128);
  const int gM = MROWS / 32;   // 32-row block tiles

  // ---- feature extract: hraw = x @ fe_w  (into xz scratch, stride 128) ----
  k_gemm<60, 64, 128><<<dim3(gM, 2), gB, 0, stream>>>(x, wfe, xz, nullptr);
  // ---- + bias, LayerNorm, ReLU -> h ----
  k_ln_relu<<<dim3(MROWS), dim3(128), 0, stream>>>(xz, fe_b, ln_g, ln_b, h);

  // ---- mamba blocks ----
  for (int l = 0; l < 2; ++l) {
    // r = rmsnorm(h) * norm_w   (r aliases the dt/y buffer, stride 128)
    k_rmsnorm<<<dim3(MROWS), dim3(128), 0, stream>>>(h, LP(l, 0), dtY);
    // xz = r @ in_proj_w
    k_gemm<128, 128, 512><<<dim3(gM, 8), gB, 0, stream>>>(dtY, wip[l], xz, nullptr);
    // xc = silu(causal_conv(xz[:, :256]))
    k_conv_silu<<<cgrid(MROWS * 256), 256, 0, stream>>>(xz, LP(l, 2), LP(l, 3), xc);
    // dbl = xc @ x_proj_w   (N padded to 80)
    k_gemm<256, 256, 80><<<dim3(gM, 2), gB, 0, stream>>>(xc, wxp[l], dbl, nullptr);
    // dt = softplus(dbl[:, :8] @ dt_proj_w + b)   (overwrites r, which is dead)
    k_dt<<<cgrid(MROWS * 256), 256, 0, stream>>>(dbl, LP(l, 5), LP(l, 6), dtY);
    // selective scan + D skip + silu(z) gate; y written in-place into dtY
    k_scan<<<dim3(NBATCH), dim3(D_INNER), 0, stream>>>(dtY, xc, dbl, xz,
                                                       LP(l, 7), LP(l, 8),
                                                       LP(l, 10), LP(l, 11));
    // h = h + y @ out_proj_w
    k_gemm<256, 256, 128><<<dim3(gM, 2), gB, 0, stream>>>(dtY, wop[l], h, h);
  }

  // ---- final rmsnorm (in-place on d_out) ----
  k_rmsnorm<<<dim3(MROWS), dim3(128), 0, stream>>>(h, final_norm_w, h);
}
